// DENetEncoder_48945447305228
// MI455X (gfx1250) — matrix operand order repaired and checked by reading, not compiled, not run on hardware
//
#include <hip/hip_runtime.h>
#include <hip/hip_bf16.h>
#include <math.h>

// Problem constants (from reference)
#define NPTS 16384
#define BB   8
#define CC   64
#define KK   16
#define MDIR 32
#define HHC  32
#define HIDC 128

typedef __attribute__((ext_vector_type(16))) __bf16 v16bf;
typedef __attribute__((ext_vector_type(8)))  __bf16 v8bf;
typedef __attribute__((ext_vector_type(8)))  float  v8f;
typedef __attribute__((ext_vector_type(4)))  unsigned int u32x4;
typedef __attribute__((ext_vector_type(8)))  int i32x8;
typedef __attribute__((ext_vector_type(4)))  int i32x4;

#define WBLOB0_US (3 * 4096 + 1024 + 2048)   // w1|w2|w3|de_w1|de_w2  (15360)
#define WBLOB1_US (HIDC * CC + CC * HIDC)    // mw1|mw2               (16384)

__device__ __forceinline__ unsigned short f2bf(float x) {
  union { float f; unsigned int u; } v; v.f = x;
  unsigned int r = v.u + 0x7fffu + ((v.u >> 16) & 1u);
  return (unsigned short)(r >> 16);
}

// Branchless tanh-form GELU: one v_exp_f32 + one v_rcp_f32, no exec-mask
// branches inside the WMMA dataflow (erff's libm expansion was ~80 VALU ops
// with saveexec branches per call).
__device__ __forceinline__ float gelu_fast(float x) {
  float u = 0.7978845608028654f * (x + 0.044715f * x * x * x);
  float z = fminf(fmaxf(u, -9.01f), 9.01f);        // avoid inf/inf
#if __has_builtin(__builtin_amdgcn_exp2f)
  float e = __builtin_amdgcn_exp2f(z * 2.8853900817779268f);  // e^(2z)
#else
  float e = exp2f(z * 2.8853900817779268f);
#endif
  float t = (e - 1.0f) * __frcp_rn(e + 1.0f);      // tanh(z)
  return 0.5f * x * (1.0f + t);
}

// ---- Tensor Data Mover: 1-D blob copy global -> LDS (6-arg builtin form) ---
// D# per CDNA5 ISA: group0 {count=1, lds_addr,
// global_addr, type=2}; group1 {data_size=3 (8B), tensor_dim0=tile_dim0=n8,
// 1-D tile (tile_dim1=0), stride=n8}; groups 2/3 zero (<=2D tensor).
#if __has_builtin(__builtin_amdgcn_tensor_load_to_lds)
#define HAVE_TDM 1
__device__ __forceinline__ void tdm_load_1d(const void* gptr, void* lptr,
                                            unsigned int bytes) {
  unsigned long long ga = (unsigned long long)(uintptr_t)gptr;
  unsigned int lo = (unsigned int)(uintptr_t)lptr;  // LDS aperture: addr[31:0]
  unsigned int n8 = bytes >> 3;                     // 8-byte elements
  u32x4 g0;
  g0.x = 1u;                                        // count=1 (valid), user D#
  g0.y = lo;                                        // lds_addr
  g0.z = (unsigned int)(ga & 0xffffffffu);          // global_addr[31:0]
  g0.w = (unsigned int)((ga >> 32) & 0x1ffffffu) | (2u << 30);  // [56:32]|type
  i32x8 g1;
  g1[0] = (int)(3u << 16);                          // data_size=3 (8 bytes)
  g1[1] = (int)((n8 & 0xffffu) << 16);              // tensor_dim0[15:0]
  g1[2] = (int)(((n8 >> 16) & 0xffffu) | (1u << 16)); // td0[31:16], td1=1
  g1[3] = (int)((n8 & 0xffffu) << 16);              // tile_dim0 = n8
  g1[4] = 0;                                        // tile_dim1=0, tile_dim2=0
  g1[5] = (int)n8;                                  // tensor_dim0_stride[31:0]
  g1[6] = 0;
  g1[7] = 0;
  i32x4 z4 = {0, 0, 0, 0};
  i32x8 z8 = {0, 0, 0, 0, 0, 0, 0, 0};
  __builtin_amdgcn_tensor_load_to_lds(g0, g1, z4, z4, z8, 0);
}
__device__ __forceinline__ void tdm_wait0() {
#if __has_builtin(__builtin_amdgcn_s_wait_tensorcnt)
  __builtin_amdgcn_s_wait_tensorcnt(0);
#else
  asm volatile("s_wait_tensorcnt 0x0" ::: "memory");
#endif
}
#else
#define HAVE_TDM 0
#endif

// A-fragment (16x32 bf16, M rows = per-lane `row`):
// lanes 0-15: elems 0-15 = K[k0..k0+15]
// lanes16-31: elems 0-15 = K[k0+16..k0+31]
__device__ __forceinline__ v16bf fragA(const unsigned short* base, int row,
                                       int rowlen, int k0, int lane) {
  int hs = (lane & 16) ? 16 : 0;
  const unsigned short* p = base + (size_t)row * rowlen + k0 + hs;
  v8bf lo = *(const v8bf*)(p);
  v8bf hi = *(const v8bf*)(p + 8);
  return __builtin_shufflevector(lo, hi, 0,1,2,3,4,5,6,7,8,9,10,11,12,13,14,15);
}

// B-fragment (32x16 bf16, N col = lane&15): lanes 0-15 hold K[k0..k0+15],
// lanes 16-31 hold K[k0+16..k0+31]; row-major source rows indexed by N.
__device__ __forceinline__ v16bf fragB(const unsigned short* base, int row,
                                       int rowlen, int k0, int lane) {
  const unsigned short* p = base + (size_t)row * rowlen + k0 + ((lane & 16) ? 16 : 0);
  return *(const v16bf*)(p);
}

__device__ __forceinline__ v8f wmma_bf16(v16bf a, v16bf b, v8f c) {
  return __builtin_amdgcn_wmma_f32_16x16x32_bf16(false, a, false, b, (short)0, c,
                                                 false, false);
}

// ---------------- Kernel P: pre-convert all weights to bf16 blobs -----------
__global__ __launch_bounds__(256)
void kp_prep(const float* __restrict__ w1, const float* __restrict__ w2,
             const float* __restrict__ w3, const float* __restrict__ de_w1,
             const float* __restrict__ de_w2, const float* __restrict__ mw1,
             const float* __restrict__ mw2, unsigned short* __restrict__ blob0,
             unsigned short* __restrict__ blob1) {
  const int tid = threadIdx.x;
  for (int t = tid; t < 4096; t += 256) {
    blob0[t] = f2bf(w1[t]);
    blob0[4096 + t] = f2bf(w2[t]);
    blob0[8192 + t] = f2bf(w3[t]);
  }
  for (int t = tid; t < 1024; t += 256) blob0[12288 + t] = f2bf(de_w1[t]);
  for (int t = tid; t < 2048; t += 256) blob0[13312 + t] = f2bf(de_w2[t]);
  for (int t = tid; t < HIDC * CC; t += 256) {
    blob1[t] = f2bf(mw1[t]);
    blob1[HIDC * CC + t] = f2bf(mw2[t]);
  }
}

// ---------------- Kernel 0: transpose f (B,C,N)f32 -> (B,N,C) f32 + bf16 ----
__global__ __launch_bounds__(256)
void k0_transpose(const float* __restrict__ f, float* __restrict__ fTf32,
                  unsigned short* __restrict__ fTbf) {
  __shared__ float tile[64][65];
  const int tid = threadIdx.x;
  const int b = blockIdx.y, n0 = blockIdx.x * 64;
  #pragma unroll
  for (int p = 0; p < 16; ++p) {
    int c = p * 4 + (tid >> 6);
    int n = tid & 63;
    tile[n][c] = f[((size_t)(b * CC + c)) * NPTS + n0 + n];
  }
  __syncthreads();
  const int n = tid >> 2, c0 = (tid & 3) * 16;
  const size_t off = ((size_t)(b * NPTS + n0 + n)) * CC + c0;
  #pragma unroll
  for (int i = 0; i < 16; ++i) fTf32[off + i] = tile[n][c0 + i];
  #pragma unroll
  for (int i = 0; i < 16; ++i) fTbf[off + i] = f2bf(tile[n][c0 + i]);
}

// ---------------- Kernel 1: directional encoding theta_max -> (B,N,32) bf16 -
__global__ __launch_bounds__(256)
void k1_theta(const float* __restrict__ dp, const float* __restrict__ dirs,
              unsigned short* __restrict__ thetabf) {
  __shared__ float sd[MDIR][4];
  const int tid = threadIdx.x;
  if (tid < MDIR) {
    float x = dirs[tid * 3 + 0], y = dirs[tid * 3 + 1], z = dirs[tid * 3 + 2];
    float inv = 1.0f / fmaxf(sqrtf(x * x + y * y + z * z), 1e-12f);
    sd[tid][0] = x * inv; sd[tid][1] = y * inv; sd[tid][2] = z * inv;
  }
  __syncthreads();
  const int b = blockIdx.y;
  const int n = blockIdx.x * 256 + tid;
  float px[KK], py[KK], pz[KK];
  const float* base = dp + (size_t)b * 3 * NPTS * KK + (size_t)n * KK;
  #pragma unroll
  for (int k = 0; k < KK; ++k) px[k] = base[k];
  #pragma unroll
  for (int k = 0; k < KK; ++k) py[k] = base[(size_t)NPTS * KK + k];
  #pragma unroll
  for (int k = 0; k < KK; ++k) pz[k] = base[2 * (size_t)NPTS * KK + k];
  #pragma unroll
  for (int k = 0; k < KK; ++k) {
    float inv = 1.0f /
        fmaxf(sqrtf(px[k] * px[k] + py[k] * py[k] + pz[k] * pz[k]), 1e-12f);
    px[k] *= inv; py[k] *= inv; pz[k] *= inv;
  }
  unsigned short outv[MDIR];
  #pragma unroll
  for (int m = 0; m < MDIR; ++m) {
    float vx = sd[m][0], vy = sd[m][1], vz = sd[m][2];
    float mx = -3.4e38f;
    #pragma unroll
    for (int k = 0; k < KK; ++k)
      mx = fmaxf(mx, vx * px[k] + vy * py[k] + vz * pz[k]);
    outv[m] = f2bf(mx);
  }
  unsigned short* dst = thetabf + ((size_t)(b * NPTS + n)) * MDIR;
  #pragma unroll
  for (int m = 0; m < MDIR; ++m) dst[m] = outv[m];
}

// ---------------- Kernel 2: WMMA front (pe chain + w1/w2/w3) ----------------
// outputs (B,N,C) f32: g1 = w1@f, g2 = w2@f,
// aux = f + pe + scale*(w3@f + b2 + b3 - g1) + shift
__global__ __launch_bounds__(256)
void k2_front(const unsigned short* __restrict__ thetabf,
              const unsigned short* __restrict__ fTbf,
              const float* __restrict__ fTf32,
              const unsigned short* __restrict__ wblob,
              const float* __restrict__ de_g1,
              const float* __restrict__ de_b1, const float* __restrict__ de_m1,
              const float* __restrict__ de_v1,
              const float* __restrict__ de_bias2,
              const float* __restrict__ b2, const float* __restrict__ b3,
              const float* __restrict__ agg_g, const float* __restrict__ agg_b,
              const float* __restrict__ agg_m, const float* __restrict__ agg_v,
              float* __restrict__ g1o, float* __restrict__ g2o,
              float* __restrict__ auxo) {
  __shared__ __attribute__((aligned(64))) unsigned short wl[WBLOB0_US];
  __shared__ float ps[4 * CC + 2 * HHC];
  __shared__ unsigned short hbuf[8][16][HHC];            // per-wave gelu(h)

  const int tid = threadIdx.x, lane = tid & 31, wv = tid >> 5;
  const int b = blockIdx.y;
  const int n0 = blockIdx.x * 128 + wv * 16;

#if HAVE_TDM
  if (wv == 0) {                         // one wave issues the DMA (EXEC ignored)
    tdm_load_1d(wblob, wl, WBLOB0_US * 2);
    tdm_wait0();
  }
#else
  for (int t = tid; t < WBLOB0_US; t += 256) wl[t] = wblob[t];
#endif
  if (tid < CC) {
    float sc = agg_g[tid] * rsqrtf(agg_v[tid] + 1e-5f);
    ps[tid] = sc;
    ps[CC + tid] = agg_b[tid] - agg_m[tid] * sc;
    ps[2 * CC + tid] = b2[tid] + b3[tid];
    ps[3 * CC + tid] = de_bias2[tid];
  }
  if (tid < HHC) {
    float sc = de_g1[tid] * rsqrtf(de_v1[tid] + 1e-5f);
    ps[4 * CC + tid] = sc;
    ps[4 * CC + HHC + tid] = de_b1[tid] - de_m1[tid] * sc;
  }
  __syncthreads();

  const int myn = n0 + (lane & 15);
  const int nb = (lane & 16) ? 8 : 0;

  // --- DE GEMM1: h = de_w1(32x32) @ theta ---
  const unsigned short* thbase = thetabf + (size_t)b * NPTS * MDIR;
  v16bf thA = fragA(thbase, myn, MDIR, 0, lane);
  v8f hacc[2];
  #pragma unroll
  for (int nt = 0; nt < 2; ++nt) {
    v8f z = {0.f, 0.f, 0.f, 0.f, 0.f, 0.f, 0.f, 0.f};
    v16bf bw = fragB(wl + 12288, nt * 16 + (lane & 15), MDIR, 0, lane);
    hacc[nt] = wmma_bf16(thA, bw, z);
  }
  // bn + gelu, stage to LDS (transpose D -> A layout for next GEMM)
  #pragma unroll
  for (int nt = 0; nt < 2; ++nt) {
    int ch = nt * 16 + (lane & 15);
    float sc = ps[4 * CC + ch], sh = ps[4 * CC + HHC + ch];
    #pragma unroll
    for (int r = 0; r < 8; ++r) {
      float x = hacc[nt][r] * sc + sh;
      hbuf[wv][r + nb][ch] = f2bf(gelu_fast(x));
    }
  }
  // --- DE GEMM2: pe = de_w2(64x32) @ h ---
  v16bf hA = fragA(&hbuf[wv][0][0], lane & 15, HHC, 0, lane);
  v8f pe[4];
  #pragma unroll
  for (int nt = 0; nt < 4; ++nt) {
    v8f z = {0.f, 0.f, 0.f, 0.f, 0.f, 0.f, 0.f, 0.f};
    v16bf bw = fragB(wl + 13312, nt * 16 + (lane & 15), HHC, 0, lane);
    pe[nt] = wmma_bf16(hA, bw, z);
  }

  // --- f GEMMs: g1/g2/g3 = w{1,2,3}(64x64) @ f ---
  const unsigned short* fbase = fTbf + (size_t)b * NPTS * CC;
  v16bf fA0 = fragA(fbase, myn, CC, 0, lane);
  v16bf fA1 = fragA(fbase, myn, CC, 32, lane);
  v8f a1[4], a2[4], a3[4];
  #pragma unroll
  for (int nt = 0; nt < 4; ++nt) {
    v8f z = {0.f, 0.f, 0.f, 0.f, 0.f, 0.f, 0.f, 0.f};
    a1[nt] = z; a2[nt] = z; a3[nt] = z;
    int orow = nt * 16 + (lane & 15);
    #pragma unroll
    for (int ks = 0; ks < 2; ++ks) {
      v16bf af = ks ? fA1 : fA0;
      a1[nt] = wmma_bf16(af, fragB(wl + 0,    orow, CC, ks * 32, lane), a1[nt]);
      a2[nt] = wmma_bf16(af, fragB(wl + 4096, orow, CC, ks * 32, lane), a2[nt]);
      a3[nt] = wmma_bf16(af, fragB(wl + 8192, orow, CC, ks * 32, lane), a3[nt]);
    }
  }

  // --- epilogue (coalesced (B,N,C) stores: lanes sweep channels) ---
  #pragma unroll
  for (int nt = 0; nt < 4; ++nt) {
    int c = nt * 16 + (lane & 15);
    float sc = ps[c], sh = ps[CC + c], cb = ps[2 * CC + c], db = ps[3 * CC + c];
    #pragma unroll
    for (int r = 0; r < 8; ++r) {
      int node = n0 + r + nb;
      size_t off = ((size_t)(b * NPTS + node)) * CC + c;
      float g1v = a1[nt][r];
      float aux = fTf32[off] + (pe[nt][r] + db) +
                  sc * (a3[nt][r] + cb - g1v) + sh;
      g1o[off] = g1v;
      g2o[off] = a2[nt][r];
      auxo[off] = aux;
    }
  }
}

// ---------------- Kernel 3: knn gather + max + combine ----------------------
__global__ __launch_bounds__(256)
void k3_gather(const int* __restrict__ qidx, const float* __restrict__ g1,
               const float* __restrict__ g2, const float* __restrict__ aux,
               const float* __restrict__ agg_g, const float* __restrict__ agg_v,
               unsigned short* __restrict__ fmidbf,
               float* __restrict__ fmidf32) {
  __shared__ int sidx[4][KK];
  const int tid = threadIdx.x;
  const int b = blockIdx.y;
  const int nb4 = blockIdx.x * 4;
  if (tid < 64)
    sidx[tid >> 4][tid & 15] =
        qidx[((size_t)(b * NPTS + nb4 + (tid >> 4))) * KK + (tid & 15)];
  __syncthreads();
  const int nl = tid >> 6, c = tid & 63;
  const int n = nb4 + nl;
  const float* g1b = g1 + (size_t)b * NPTS * CC;
  const float* g2b = g2 + (size_t)b * NPTS * CC;
  float m1 = -3.4e38f, m2 = -3.4e38f;
  #pragma unroll
  for (int k = 0; k < KK; ++k) {
    int j = sidx[nl][k];
    if (k + 1 < KK) {
      int j2 = sidx[nl][k + 1];
      __builtin_prefetch(g1b + (size_t)j2 * CC + c, 0, 1);
      __builtin_prefetch(g2b + (size_t)j2 * CC + c, 0, 1);
    }
    m1 = fmaxf(m1, g1b[(size_t)j * CC + c]);
    m2 = fmaxf(m2, g2b[(size_t)j * CC + c]);
  }
  float sc = agg_g[c] * rsqrtf(agg_v[c] + 1e-5f);
  size_t off = ((size_t)(b * NPTS + n)) * CC + c;
  float v = aux[off] + sc * (m1 + m2);
  fmidf32[off] = v;
  fmidbf[off] = f2bf(v);
}

// ---------------- Kernel 4: WMMA MLP + residual -> out (B,C,N) --------------
__global__ __launch_bounds__(256)
void k4_mlp(const unsigned short* __restrict__ fmidbf,
            const float* __restrict__ fmidf32,
            const unsigned short* __restrict__ wblob,
            const float* __restrict__ mg, const float* __restrict__ mbp,
            const float* __restrict__ mmp, const float* __restrict__ mvp,
            float* __restrict__ out) {
  __shared__ __attribute__((aligned(64))) unsigned short wl[WBLOB1_US];
  __shared__ float ps[2 * HIDC];
  __shared__ unsigned short h2buf[8][16][HIDC];

  const int tid = threadIdx.x, lane = tid & 31, wv = tid >> 5;
  const int b = blockIdx.y;
  const int n0 = blockIdx.x * 128 + wv * 16;

#if HAVE_TDM
  if (wv == 0) {
    tdm_load_1d(wblob, wl, WBLOB1_US * 2);
    tdm_wait0();
  }
#else
  for (int t = tid; t < WBLOB1_US; t += 256) wl[t] = wblob[t];
#endif
  if (tid < HIDC) {
    float sc = mg[tid] * rsqrtf(mvp[tid] + 1e-5f);
    ps[tid] = sc;
    ps[HIDC + tid] = mbp[tid] - mmp[tid] * sc;
  }
  __syncthreads();

  const unsigned short* wl1 = wl;              // mw1 rows (128 x 64)
  const unsigned short* wl2 = wl + HIDC * CC;  // mw2 rows (64 x 128)
  const int myn = n0 + (lane & 15);
  const int nb = (lane & 16) ? 8 : 0;
  const unsigned short* fbase = fmidbf + (size_t)b * NPTS * CC;
  v16bf fA0 = fragA(fbase, myn, CC, 0, lane);
  v16bf fA1 = fragA(fbase, myn, CC, 32, lane);

  // GEMM1: h = mw1(128x64) @ fmid  (acts as A, weights as B)
  v8f h[8];
  #pragma unroll
  for (int nt = 0; nt < 8; ++nt) {
    v8f z = {0.f, 0.f, 0.f, 0.f, 0.f, 0.f, 0.f, 0.f};
    int orow = nt * 16 + (lane & 15);
    z = wmma_bf16(fA0, fragB(wl1, orow, CC, 0, lane), z);
    h[nt] = wmma_bf16(fA1, fragB(wl1, orow, CC, 32, lane), z);
  }
  // bn + gelu, stage (node, hid) rows
  #pragma unroll
  for (int nt = 0; nt < 8; ++nt) {
    int ch = nt * 16 + (lane & 15);
    float sc = ps[ch], sh = ps[HIDC + ch];
    #pragma unroll
    for (int r = 0; r < 8; ++r) {
      float x = h[nt][r] * sc + sh;
      h2buf[wv][r + nb][ch] = f2bf(gelu_fast(x));
    }
  }
  // GEMM2: o = mw2(64x128) @ h2  (weights as A, h2 as B -> lanes sweep nodes)
  v8f o[4];
  #pragma unroll
  for (int ct = 0; ct < 4; ++ct) {
    v8f z = {0.f, 0.f, 0.f, 0.f, 0.f, 0.f, 0.f, 0.f};
    o[ct] = z;
  }
  #pragma unroll
  for (int ks = 0; ks < 4; ++ks) {
    v16bf bh = fragB(&h2buf[wv][0][0], lane & 15, HIDC, ks * 32, lane);
    #pragma unroll
    for (int ct = 0; ct < 4; ++ct) {
      v16bf aw = fragA(wl2, ct * 16 + (lane & 15), HIDC, ks * 32, lane);
      o[ct] = wmma_bf16(aw, bh, o[ct]);
    }
  }
  // epilogue: out[b][c][n] = fmid + o  (lanes sweep nodes -> coalesced store)
  const int node = n0 + (lane & 15);
  #pragma unroll
  for (int ct = 0; ct < 4; ++ct) {
    #pragma unroll
    for (int r = 0; r < 8; ++r) {
      int c = ct * 16 + r + nb;
      float res = fmidf32[((size_t)(b * NPTS + node)) * CC + c];
      out[((size_t)(b * CC + c)) * NPTS + node] = o[ct][r] + res;
    }
  }
}

extern "C" void kernel_launch(void* const* d_in, const int* in_sizes, int n_in,
                              void* d_out, int out_size, void* d_ws,
                              size_t ws_size, hipStream_t stream) {
  (void)in_sizes; (void)n_in; (void)out_size; (void)ws_size;
  const float* f        = (const float*)d_in[0];
  const float* dp       = (const float*)d_in[1];
  const int*   qidx     = (const int*)d_in[2];
  const float* dirs     = (const float*)d_in[3];
  const float* de_w1    = (const float*)d_in[4];
  const float* de_g1    = (const float*)d_in[5];
  const float* de_b1    = (const float*)d_in[6];
  const float* de_m1    = (const float*)d_in[7];
  const float* de_v1    = (const float*)d_in[8];
  const float* de_w2    = (const float*)d_in[9];
  const float* de_bias2 = (const float*)d_in[10];
  const float* w1       = (const float*)d_in[11];
  /* b1 (d_in[12]) cancels in edge-max algebra */
  const float* w2       = (const float*)d_in[13];
  const float* b2       = (const float*)d_in[14];
  const float* w3       = (const float*)d_in[15];
  const float* b3       = (const float*)d_in[16];
  const float* agg_g    = (const float*)d_in[17];
  const float* agg_b    = (const float*)d_in[18];
  const float* agg_m    = (const float*)d_in[19];
  const float* agg_v    = (const float*)d_in[20];
  const float* mw1      = (const float*)d_in[21];
  const float* mg       = (const float*)d_in[22];
  const float* mb       = (const float*)d_in[23];
  const float* mm       = (const float*)d_in[24];
  const float* mv       = (const float*)d_in[25];
  const float* mw2      = (const float*)d_in[26];

  char* ws = (char*)d_ws;
  const size_t szBNCbf = (size_t)BB * NPTS * CC * 2;    // 16 MB
  const size_t szBNCf  = (size_t)BB * NPTS * CC * 4;    // 32 MB
  const size_t szTheta = (size_t)BB * NPTS * MDIR * 2;  //  8 MB
  // fTbf/fTf32 are consumed by k2, then reused as fmid buffers by k3/k4.
  unsigned short* fTbf    = (unsigned short*)(ws);
  float*          fTf32   = (float*)(ws + szBNCbf);
  unsigned short* thetabf = (unsigned short*)(ws + szBNCbf + szBNCf);
  float* g1  = (float*)(ws + szBNCbf + szBNCf + szTheta);
  float* g2  = (float*)(ws + szBNCbf + szBNCf + szTheta + szBNCf);
  float* aux = (float*)(ws + szBNCbf + szBNCf + szTheta + 2 * szBNCf);
  unsigned short* blob0 =
      (unsigned short*)(ws + szBNCbf + szBNCf + szTheta + 3 * szBNCf);
  unsigned short* blob1 = blob0 + WBLOB0_US;

  kp_prep<<<dim3(1), 256, 0, stream>>>(w1, w2, w3, de_w1, de_w2, mw1, mw2,
                                       blob0, blob1);
  k0_transpose<<<dim3(NPTS / 64, BB), 256, 0, stream>>>(f, fTf32, fTbf);
  k1_theta<<<dim3(NPTS / 256, BB), 256, 0, stream>>>(dp, dirs, thetabf);
  k2_front<<<dim3(NPTS / 128, BB), 256, 0, stream>>>(
      thetabf, fTbf, fTf32, blob0, de_g1, de_b1, de_m1, de_v1, de_bias2, b2, b3,
      agg_g, agg_b, agg_m, agg_v, g1, g2, aux);
  k3_gather<<<dim3(NPTS / 4, BB), 256, 0, stream>>>(qidx, g1, g2, aux, agg_g,
                                                    agg_v, fTbf, fTf32);
  k4_mlp<<<dim3(NPTS / 128, BB), 256, 0, stream>>>(fTbf, fTf32, blob1, mg, mb,
                                                   mm, mv, (float*)d_out);
}
